// Rwkv7SelfAttention_89730456748340
// MI455X (gfx1250) — matrix operand order repaired and checked by reading, not compiled, not run on hardware
//
#include <hip/hip_runtime.h>

#define SEQ_T 4096
#define HID   2048
#define NH    32
#define HSZ   64
#define W_SCALE_C (-0.6065306597126334f)

typedef __bf16 bf16;
typedef __bf16 v16bf __attribute__((ext_vector_type(16)));
typedef __bf16 v8bf  __attribute__((ext_vector_type(8)));
typedef float  v8f   __attribute__((ext_vector_type(8)));

// ---------------------------------------------------------------------------
// LayerNorm over hidden (ln_1), one block per row
// ---------------------------------------------------------------------------
__global__ __launch_bounds__(256) void k_ln1(const float* __restrict__ x,
                                             const float* __restrict__ w,
                                             const float* __restrict__ b,
                                             float* __restrict__ xn) {
  const int t = blockIdx.x, tid = threadIdx.x;
  __shared__ float red[256];
  const float* row = x + (size_t)t * HID;
  float s = 0.f;
  for (int j = tid; j < HID; j += 256) s += row[j];
  red[tid] = s; __syncthreads();
  for (int o = 128; o > 0; o >>= 1) { if (tid < o) red[tid] += red[tid + o]; __syncthreads(); }
  const float mean = red[0] * (1.f / HID);
  __syncthreads();
  float vs = 0.f;
  for (int j = tid; j < HID; j += 256) { float d = row[j] - mean; vs += d * d; }
  red[tid] = vs; __syncthreads();
  for (int o = 128; o > 0; o >>= 1) { if (tid < o) red[tid] += red[tid + o]; __syncthreads(); }
  const float inv = rsqrtf(red[0] * (1.f / HID) + 1e-5f);
  for (int j = tid; j < HID; j += 256)
    xn[(size_t)t * HID + j] = (row[j] - mean) * inv * w[j] + b[j];
}

// ---------------------------------------------------------------------------
// Token shift -> six bf16 mixed streams
// ---------------------------------------------------------------------------
__global__ __launch_bounds__(256) void k_shift(
    const float* __restrict__ xn, const float* __restrict__ state1,
    const float* __restrict__ mr, const float* __restrict__ mw,
    const float* __restrict__ mk, const float* __restrict__ mv,
    const float* __restrict__ ma, const float* __restrict__ mg,
    bf16* __restrict__ xr, bf16* __restrict__ xw, bf16* __restrict__ xk,
    bf16* __restrict__ xv, bf16* __restrict__ xa, bf16* __restrict__ xg) {
  const int t = blockIdx.x;
  for (int j = threadIdx.x; j < HID; j += 256) {
    const size_t idx = (size_t)t * HID + j;
    const float cur = xn[idx];
    const float prev = t ? xn[idx - HID] : state1[j];
    const float sx = prev - cur;
    xr[idx] = (bf16)(cur + sx * mr[j]);
    xw[idx] = (bf16)(cur + sx * mw[j]);
    xk[idx] = (bf16)(cur + sx * mk[j]);
    xv[idx] = (bf16)(cur + sx * mv[j]);
    xa[idx] = (bf16)(cur + sx * ma[j]);
    xg[idx] = (bf16)(cur + sx * mg[j]);
  }
}

// ---------------------------------------------------------------------------
// f32 -> bf16 convert, optional activation (0=identity, 1=tanh, 2=sigmoid)
// ---------------------------------------------------------------------------
__global__ __launch_bounds__(256) void k_act_bf(const float* __restrict__ s,
                                                bf16* __restrict__ d,
                                                size_t n, int mode) {
  size_t id = (size_t)blockIdx.x * 256 + threadIdx.x;
  if (id >= n) return;
  float v = s[id];
  if (mode == 1) v = tanhf(v);
  else if (mode == 2) v = 1.f / (1.f + __expf(-v));
  d[id] = (bf16)v;
}

// transpose [rows, cols] f32 -> [cols, rows] bf16
__global__ __launch_bounds__(256) void k_transpose_bf(const float* __restrict__ src,
                                                      bf16* __restrict__ dst,
                                                      int rows, int cols) {
  size_t id = (size_t)blockIdx.x * 256 + threadIdx.x;
  size_t n = (size_t)rows * cols;
  if (id >= n) return;
  int r = (int)(id / cols), c = (int)(id % cols);
  dst[(size_t)c * rows + r] = (bf16)src[id];
}

// ---------------------------------------------------------------------------
// WMMA bf16 GEMM:  C[M,N] = A[M,K] * B[N,K]^T  (f32 accumulate)
// block = 256 threads (8 waves as 4x2), block tile 256x128, wave tile 64x64
// (4x4 fragments -> 16 WMMA per 16 b128 loads per 32-K step)
// Fragment layouts per CDNA5 ISA 7.12.2 (16-bit A 16x32, B 32x16, f32 C 16x16)
// ---------------------------------------------------------------------------
__global__ __launch_bounds__(256) void k_gemm_bf16(
    const bf16* __restrict__ A, const bf16* __restrict__ B,
    float* __restrict__ C, int M, int N, int K) {
  const int lane = threadIdx.x & 31;
  const int wave = threadIdx.x >> 5;
  const int wy = wave >> 1, wx = wave & 1;
  const int wm = blockIdx.x * 256 + wy * 64;
  const int wn = blockIdx.y * 128 + wx * 64;
  const int lr = lane & 15;
  const int hs = lane >> 4;  // half-select (lanes 16..31)

  v8f acc[4][4];
#pragma unroll
  for (int i = 0; i < 4; i++)
#pragma unroll
    for (int j = 0; j < 4; j++)
#pragma unroll
      for (int e = 0; e < 8; e++) acc[i][j][e] = 0.f;

  int arow[4], brow[4];
#pragma unroll
  for (int i = 0; i < 4; i++) { int r = wm + i * 16 + lr; arow[i] = r < M ? r : M - 1; }
#pragma unroll
  for (int j = 0; j < 4; j++) { int r = wn + j * 16 + lr; brow[j] = r < N ? r : N - 1; }

  for (int kk = 0; kk < K; kk += 32) {
    v16bf af[4], bfr[4];
    const int ab = kk + hs * 8;   // A: lanes>=16 hold K rows 8..15 / 24..31
#pragma unroll
    for (int i = 0; i < 4; i++) {
      const bf16* ap = A + (size_t)arow[i] * K + ab;
      v8bf lo = *(const v8bf*)ap;           // K = ab .. ab+7
      v8bf hi = *(const v8bf*)(ap + 16);    // K = ab+16 .. ab+23
#pragma unroll
      for (int e = 0; e < 8; e++) { af[i][e] = lo[e]; af[i][e + 8] = hi[e]; }
    }
    const int bb = kk + hs * 8;   // B: lanes>=16 hold K rows 8..15 / 24..31
#pragma unroll
    for (int j = 0; j < 4; j++) {
      const bf16* bp = B + (size_t)brow[j] * K + bb;
      v8bf lo = *(const v8bf*)bp;           // K = bb .. bb+7
      v8bf hi = *(const v8bf*)(bp + 16);    // K = bb+16 .. bb+23
#pragma unroll
      for (int e = 0; e < 8; e++) { bfr[j][e] = lo[e]; bfr[j][e + 8] = hi[e]; }
    }
    // Speculative RT prefetch into GL2 for the next K step (safe if OOB:
    // speculative prefetches with invalid addresses are silently dropped).
    __builtin_prefetch(A + (size_t)arow[0] * K + kk + 32, 0, 0);
    __builtin_prefetch(B + (size_t)brow[0] * K + kk + 32, 0, 0);
#pragma unroll
    for (int i = 0; i < 4; i++)
#pragma unroll
      for (int j = 0; j < 4; j++)
        acc[i][j] = __builtin_amdgcn_wmma_f32_16x16x32_bf16(
            false, af[i], false, bfr[j], (short)0, acc[i][j], false, false);
  }

#pragma unroll
  for (int i = 0; i < 4; i++) {
    const int rbase = wm + i * 16 + hs * 8;
#pragma unroll
    for (int j = 0; j < 4; j++) {
      const int col = wn + j * 16 + lr;
      if (col < N) {
#pragma unroll
        for (int e = 0; e < 8; e++) {
          const int row = rbase + e;
          if (row < M) C[(size_t)row * N + col] = acc[i][j][e];
        }
      }
    }
  }
}

// ---------------------------------------------------------------------------
// Elementwise prep: kk l2-norm per head, ah/bh, k/v in-place update, decay.
// grid (T, NH), block 64 (one head-row per block)
// ---------------------------------------------------------------------------
__global__ __launch_bounds__(64) void k_prep(
    float* __restrict__ k, float* __restrict__ v,
    const float* __restrict__ v_first,
    float* __restrict__ w2o,                   // in: xw mlp out, out: decay
    const float* __restrict__ a2o, const float* __restrict__ v2o,
    const float* __restrict__ w0, const float* __restrict__ a0,
    const float* __restrict__ v0,
    const float* __restrict__ k_k, const float* __restrict__ k_a,
    float* __restrict__ ah, float* __restrict__ bh) {
  const int t = blockIdx.x, h = blockIdx.y, j = threadIdx.x;
  const int hid = h * HSZ + j;
  const size_t idx = (size_t)t * HID + hid;
  __shared__ float red[64];

  const float kraw = k[idx];
  const float kk0 = kraw * k_k[hid];
  red[j] = kk0 * kk0; __syncthreads();
  for (int o = 32; o > 0; o >>= 1) { if (j < o) red[j] += red[j + o]; __syncthreads(); }
  const float inv = 1.f / (sqrtf(red[0]) + 1e-12f);
  const float kkn = kk0 * inv;

  const float asig = 1.f / (1.f + __expf(-(a2o[idx] + a0[hid])));
  ah[idx] = -kkn;
  bh[idx] = kkn * asig;
  k[idx] = kraw * (1.f + (asig - 1.f) * k_a[hid]);

  const float wsig = 1.f / (1.f + __expf(-(w2o[idx] + w0[hid])));
  w2o[idx] = __expf(W_SCALE_C * wsig);

  const float vg = 1.f / (1.f + __expf(-(v2o[idx] + v0[hid])));
  const float vv = v[idx];
  v[idx] = vv + (v_first[idx] - vv) * vg;
}

// ---------------------------------------------------------------------------
// Sequential WKV scan. One block (64 threads) per head; thread i owns state
// row S[i, 0..63] in VGPRs. Per-step 64-vectors broadcast via LDS.
// ---------------------------------------------------------------------------
__global__ __launch_bounds__(64) void k_scan(
    const float* __restrict__ r, const float* __restrict__ decay,
    const float* __restrict__ k, const float* __restrict__ v,
    const float* __restrict__ ah, const float* __restrict__ bh,
    const float* __restrict__ state2, float* __restrict__ xo) {
  const int h = blockIdx.x;
  const int i = threadIdx.x;
  __shared__ float rs[64], ds[64], ks[64], as_[64], bs[64];

  float s[64];
  const float* s0 = state2 + ((size_t)h * HSZ + i) * HSZ;
#pragma unroll
  for (int j = 0; j < 64; j++) s[j] = s0[j];

  for (int t = 0; t < SEQ_T; t++) {
    const size_t base = (size_t)t * HID + h * HSZ;
    rs[i] = r[base + i];  ds[i] = decay[base + i]; ks[i] = k[base + i];
    as_[i] = ah[base + i]; bs[i] = bh[base + i];
    const float vi = v[base + i];
    __syncthreads();
    float sa = 0.f;
#pragma unroll
    for (int j = 0; j < 64; j++) sa += s[j] * as_[j];
    float o = 0.f;
#pragma unroll
    for (int j = 0; j < 64; j++) {
      const float sv = s[j] * ds[j] + sa * bs[j] + vi * ks[j];
      s[j] = sv;
      o += sv * rs[j];
    }
    xo[base + i] = o;
    __syncthreads();
  }
}

// ---------------------------------------------------------------------------
// Per-head LN (eps=0.00064) + affine + rkv + gate -> bf16 y.  grid (T, NH)
// ---------------------------------------------------------------------------
__global__ __launch_bounds__(64) void k_post(
    const float* __restrict__ xo, const float* __restrict__ r,
    const float* __restrict__ k, const float* __restrict__ v,
    const float* __restrict__ r_k, const float* __restrict__ g,
    const float* __restrict__ ln_x_w, const float* __restrict__ ln_x_b,
    bf16* __restrict__ y) {
  const int t = blockIdx.x, h = blockIdx.y, j = threadIdx.x;
  const int hid = h * HSZ + j;
  const size_t idx = (size_t)t * HID + hid;
  __shared__ float red[64];

  const float xv = xo[idx];
  red[j] = xv; __syncthreads();
  for (int o = 32; o > 0; o >>= 1) { if (j < o) red[j] += red[j + o]; __syncthreads(); }
  const float mean = red[0] * (1.f / 64.f);
  __syncthreads();
  const float d = xv - mean;
  red[j] = d * d; __syncthreads();
  for (int o = 32; o > 0; o >>= 1) { if (j < o) red[j] += red[j + o]; __syncthreads(); }
  const float var = red[0] * (1.f / 64.f);
  __syncthreads();
  float xln = d * rsqrtf(var + 0.00064f);
  xln = xln * ln_x_w[hid] + ln_x_b[hid];

  red[j] = r[idx] * k[idx] * r_k[hid]; __syncthreads();
  for (int o = 32; o > 0; o >>= 1) { if (j < o) red[j] += red[j + o]; __syncthreads(); }
  const float rkv = red[0] * v[idx];

  y[idx] = (bf16)((xln + rkv) * g[idx]);
}

__global__ __launch_bounds__(256) void k_addout(const float* __restrict__ x,
                                                const float* __restrict__ t,
                                                float* __restrict__ out, size_t n) {
  size_t id = (size_t)blockIdx.x * 256 + threadIdx.x;
  if (id < n) out[id] = x[id] + t[id];
}

// ---------------------------------------------------------------------------
extern "C" void kernel_launch(void* const* d_in, const int* in_sizes, int n_in,
                              void* d_out, int out_size, void* d_ws, size_t ws_size,
                              hipStream_t stream) {
  (void)in_sizes; (void)n_in; (void)out_size; (void)ws_size;
  const float* x      = (const float*)d_in[0];
  const float* state1 = (const float*)d_in[1];
  const float* state2 = (const float*)d_in[2];
  const float* v_first= (const float*)d_in[3];
  const float* ln1_w  = (const float*)d_in[4];
  const float* ln1_b  = (const float*)d_in[5];
  const float* x_r = (const float*)d_in[6];
  const float* x_w = (const float*)d_in[7];
  const float* x_k = (const float*)d_in[8];
  const float* x_v = (const float*)d_in[9];
  const float* x_a = (const float*)d_in[10];
  const float* x_g = (const float*)d_in[11];
  const float* Wr = (const float*)d_in[12];
  const float* Wk = (const float*)d_in[13];
  const float* Wv = (const float*)d_in[14];
  const float* Wo = (const float*)d_in[15];
  const float* w1 = (const float*)d_in[16];
  const float* w2 = (const float*)d_in[17];
  const float* w0 = (const float*)d_in[18];
  const float* a1 = (const float*)d_in[19];
  const float* a2 = (const float*)d_in[20];
  const float* a0 = (const float*)d_in[21];
  const float* v1 = (const float*)d_in[22];
  const float* v2 = (const float*)d_in[23];
  const float* v0 = (const float*)d_in[24];
  const float* g1 = (const float*)d_in[25];
  const float* g2 = (const float*)d_in[26];
  const float* k_k = (const float*)d_in[27];
  const float* k_a = (const float*)d_in[28];
  const float* r_k = (const float*)d_in[29];
  const float* ln_x_w = (const float*)d_in[30];
  const float* ln_x_b = (const float*)d_in[31];
  float* out = (float*)d_out;

  const size_t TH = (size_t)SEQ_T * HID;
  size_t off = 0;
  auto alloc = [&](size_t bytes) -> void* {
    void* p = (void*)((char*)d_ws + off);
    off += (bytes + 255) & ~(size_t)255;
    return p;
  };

  float* xn   = (float*)alloc(TH * 4);   // later reused as final GEMM temp
  bf16* xr_b = (bf16*)alloc(TH * 2);
  bf16* xw_b = (bf16*)alloc(TH * 2);
  bf16* xk_b = (bf16*)alloc(TH * 2);
  bf16* xv_b = (bf16*)alloc(TH * 2);
  bf16* xa_b = (bf16*)alloc(TH * 2);
  bf16* xg_b = (bf16*)alloc(TH * 2);
  bf16* Wr_b = (bf16*)alloc((size_t)HID * HID * 2);
  bf16* Wk_b = (bf16*)alloc((size_t)HID * HID * 2);
  bf16* Wv_b = (bf16*)alloc((size_t)HID * HID * 2);
  bf16* Wo_b = (bf16*)alloc((size_t)HID * HID * 2);
  bf16* w1T = (bf16*)alloc((size_t)64 * HID * 2);
  bf16* w2T = (bf16*)alloc((size_t)HID * 64 * 2);
  bf16* a1T = (bf16*)alloc((size_t)64 * HID * 2);
  bf16* a2T = (bf16*)alloc((size_t)HID * 64 * 2);
  bf16* v1T = (bf16*)alloc((size_t)32 * HID * 2);
  bf16* v2T = (bf16*)alloc((size_t)HID * 32 * 2);
  bf16* g1T = (bf16*)alloc((size_t)128 * HID * 2);
  bf16* g2T = (bf16*)alloc((size_t)HID * 128 * 2);
  float* r_f = (float*)alloc(TH * 4);
  float* k_f = (float*)alloc(TH * 4);
  float* v_f = (float*)alloc(TH * 4);
  float* hw  = (float*)alloc((size_t)SEQ_T * 64 * 4);
  bf16*  hw_b = (bf16*)alloc((size_t)SEQ_T * 64 * 2);
  float* ha  = (float*)alloc((size_t)SEQ_T * 64 * 4);
  bf16*  ha_b = (bf16*)alloc((size_t)SEQ_T * 64 * 2);
  float* hv  = (float*)alloc((size_t)SEQ_T * 32 * 4);
  bf16*  hv_b = (bf16*)alloc((size_t)SEQ_T * 32 * 2);
  float* hg  = (float*)alloc((size_t)SEQ_T * 128 * 4);
  bf16*  hg_b = (bf16*)alloc((size_t)SEQ_T * 128 * 2);
  float* w2o = (float*)alloc(TH * 4);    // becomes decay after k_prep
  float* a2o = (float*)alloc(TH * 4);
  float* v2o = (float*)alloc(TH * 4);
  float* g2o = (float*)alloc(TH * 4);
  float* ahb = (float*)alloc(TH * 4);
  float* bhb = (float*)alloc(TH * 4);
  float* xo  = (float*)alloc(TH * 4);
  bf16*  y_b = (bf16*)alloc(TH * 2);
  float* tmp = xn;                        // reuse xn region for Wo GEMM output

  auto eg = [](size_t n) { return dim3((unsigned)((n + 255) / 256)); };
  auto gg = [](int M, int N) { return dim3((M + 255) / 256, (N + 127) / 128); };

  // 1) ln_1 and token shift
  k_ln1<<<SEQ_T, 256, 0, stream>>>(x, ln1_w, ln1_b, xn);
  k_shift<<<SEQ_T, 256, 0, stream>>>(xn, state1, x_r, x_w, x_k, x_v, x_a, x_g,
                                     xr_b, xw_b, xk_b, xv_b, xa_b, xg_b);

  // 2) weight conversions
  const size_t WN = (size_t)HID * HID;
  k_act_bf<<<eg(WN), 256, 0, stream>>>(Wr, Wr_b, WN, 0);
  k_act_bf<<<eg(WN), 256, 0, stream>>>(Wk, Wk_b, WN, 0);
  k_act_bf<<<eg(WN), 256, 0, stream>>>(Wv, Wv_b, WN, 0);
  k_act_bf<<<eg(WN), 256, 0, stream>>>(Wo, Wo_b, WN, 0);
  k_transpose_bf<<<eg((size_t)HID * 64), 256, 0, stream>>>(w1, w1T, HID, 64);
  k_transpose_bf<<<eg((size_t)HID * 64), 256, 0, stream>>>(w2, w2T, 64, HID);
  k_transpose_bf<<<eg((size_t)HID * 64), 256, 0, stream>>>(a1, a1T, HID, 64);
  k_transpose_bf<<<eg((size_t)HID * 64), 256, 0, stream>>>(a2, a2T, 64, HID);
  k_transpose_bf<<<eg((size_t)HID * 32), 256, 0, stream>>>(v1, v1T, HID, 32);
  k_transpose_bf<<<eg((size_t)HID * 32), 256, 0, stream>>>(v2, v2T, 32, HID);
  k_transpose_bf<<<eg((size_t)HID * 128), 256, 0, stream>>>(g1, g1T, HID, 128);
  k_transpose_bf<<<eg((size_t)HID * 128), 256, 0, stream>>>(g2, g2T, 128, HID);

  // 3) big projections (WMMA)
  k_gemm_bf16<<<gg(SEQ_T, HID), 256, 0, stream>>>(xr_b, Wr_b, r_f, SEQ_T, HID, HID);
  k_gemm_bf16<<<gg(SEQ_T, HID), 256, 0, stream>>>(xk_b, Wk_b, k_f, SEQ_T, HID, HID);
  k_gemm_bf16<<<gg(SEQ_T, HID), 256, 0, stream>>>(xv_b, Wv_b, v_f, SEQ_T, HID, HID);

  // 4) low-rank MLP first stages (WMMA)
  k_gemm_bf16<<<gg(SEQ_T, 64), 256, 0, stream>>>(xw_b, w1T, hw, SEQ_T, 64, HID);
  k_gemm_bf16<<<gg(SEQ_T, 64), 256, 0, stream>>>(xa_b, a1T, ha, SEQ_T, 64, HID);
  k_gemm_bf16<<<gg(SEQ_T, 32), 256, 0, stream>>>(xv_b, v1T, hv, SEQ_T, 32, HID);
  k_gemm_bf16<<<gg(SEQ_T, 128), 256, 0, stream>>>(xg_b, g1T, hg, SEQ_T, 128, HID);

  // 5) inner activations
  k_act_bf<<<eg((size_t)SEQ_T * 64), 256, 0, stream>>>(hw, hw_b, (size_t)SEQ_T * 64, 1);  // tanh
  k_act_bf<<<eg((size_t)SEQ_T * 64), 256, 0, stream>>>(ha, ha_b, (size_t)SEQ_T * 64, 0);
  k_act_bf<<<eg((size_t)SEQ_T * 32), 256, 0, stream>>>(hv, hv_b, (size_t)SEQ_T * 32, 0);
  k_act_bf<<<eg((size_t)SEQ_T * 128), 256, 0, stream>>>(hg, hg_b, (size_t)SEQ_T * 128, 2); // sigmoid

  // 6) low-rank MLP second stages (WMMA)
  k_gemm_bf16<<<gg(SEQ_T, HID), 256, 0, stream>>>(hw_b, w2T, w2o, SEQ_T, HID, 64);
  k_gemm_bf16<<<gg(SEQ_T, HID), 256, 0, stream>>>(ha_b, a2T, a2o, SEQ_T, HID, 64);
  k_gemm_bf16<<<gg(SEQ_T, HID), 256, 0, stream>>>(hv_b, v2T, v2o, SEQ_T, HID, 32);
  k_gemm_bf16<<<gg(SEQ_T, HID), 256, 0, stream>>>(hg_b, g2T, g2o, SEQ_T, HID, 128);

  // 7) elementwise prep (decay/exp, kk norm, ah/bh, k & v in-place update)
  k_prep<<<dim3(SEQ_T, NH), 64, 0, stream>>>(k_f, v_f, v_first, w2o, a2o, v2o,
                                             w0, a0, v0, k_k, k_a, ahb, bhb);

  // 8) sequential WKV scan
  k_scan<<<NH, 64, 0, stream>>>(r_f, w2o, k_f, v_f, ahb, bhb, state2, xo);

  // 9) per-head LN + rkv + gate -> bf16
  k_post<<<dim3(SEQ_T, NH), 64, 0, stream>>>(xo, r_f, k_f, v_f, r_k, g2o,
                                             ln_x_w, ln_x_b, y_b);

  // 10) output projection (WMMA) + residual
  k_gemm_bf16<<<gg(SEQ_T, HID), 256, 0, stream>>>(y_b, Wo_b, tmp, SEQ_T, HID, HID);
  k_addout<<<eg(TH), 256, 0, stream>>>(x, tmp, out, TH);
}
